// MixGNN_56762287784200
// MI455X (gfx1250) — matrix operand order repaired and checked by reading, not compiled, not run on hardware
//
#include <hip/hip_runtime.h>
#include <hip/hip_bf16.h>
#include <stdint.h>

#define D 256
#define BN_SCALE 0.9999950000374997f  /* 1/sqrt(1+1e-5) */
#define BROW 264                       /* padded LDS row stride (bank-conflict-free) */

typedef __attribute__((ext_vector_type(16))) __bf16 v16bf;
typedef __attribute__((ext_vector_type(8)))  float  v8f;
typedef int v4i_async __attribute__((vector_size(16)));   // matches builtin param type

union Frag32B { uint4 u[2]; v16bf v; };

#if defined(__has_builtin)
#if __has_builtin(__builtin_amdgcn_global_load_async_to_lds_b128)
#define HAVE_ASYNC_LDS 1
#endif
#endif

// ---------- helpers ----------
__device__ __forceinline__ unsigned short f2bf(float f) {
    unsigned u = __float_as_uint(f);
    u += 0x7FFFu + ((u >> 16) & 1u);        // round-to-nearest-even
    return (unsigned short)(u >> 16);
}
__device__ __forceinline__ unsigned fkey(float f) {       // order-preserving float->uint
    unsigned u = __float_as_uint(f);
    return (u & 0x80000000u) ? ~u : (u | 0x80000000u);
}
__device__ __forceinline__ float unkey(unsigned k) {
    unsigned u = (k & 0x80000000u) ? (k & 0x7FFFFFFFu) : ~k;
    return __uint_as_float(u);
}

// ---------- elementwise / packing ----------
__global__ void cast_bf16_k(const float* __restrict__ src, unsigned short* __restrict__ dst, int n) {
    int i = blockIdx.x * blockDim.x + threadIdx.x;
    if (i < n) dst[i] = f2bf(src[i]);
}
// Bt[n*256+k] = bf16(B[k*256+n])  (weight transpose+cast, done once per launch)
__global__ void transpose_cast_k(const float* __restrict__ src, unsigned short* __restrict__ dst) {
    int k = threadIdx.x, n = blockIdx.x;
    dst[n * D + k] = f2bf(src[k * D + n]);
}
__global__ void zero_f32_k(float* __restrict__ p, int n) {
    int i = blockIdx.x * blockDim.x + threadIdx.x;
    if (i < n) p[i] = 0.0f;
}
__global__ void init_nodes_k(unsigned* __restrict__ mkey, float* __restrict__ ssum,
                             float* __restrict__ deg, int n) {
    int i = blockIdx.x * blockDim.x + threadIdx.x;
    if (i < n) { mkey[i] = 0x007FFFFFu; /* key(-inf) */ ssum[i] = 0.0f; deg[i] = 0.0f; }
}

// ---------- WMMA GEMM:  C[M,256] = A[M,256] @ B[256,256] (+bias) ----------
// A: bf16 row-major.  Bt: bf16, Bt[n*256+k] = B[k][n].  C: f32.
// Block = 128 threads (4 waves). Each block: 128 rows x 64 cols.
// B-strip (64 cols x 256 K, 32KB) staged into LDS once per block (async-to-LDS).
// Each wave: 2 row-tiles x 64 cols = 64 WMMAs.
__global__ __launch_bounds__(128)
void gemm_bf16_wmma_k(const unsigned short* __restrict__ A,
                      const unsigned short* __restrict__ Bt,
                      const float* __restrict__ bias,
                      float* __restrict__ C, int ntiles)
{
    __shared__ unsigned short ldsB[64 * BROW];

    const int lane = threadIdx.x & 31;
    const int wave = threadIdx.x >> 5;
    const int n0   = blockIdx.y << 6;

    // ---- stage B strip (64 rows of Bt, 512B each) into LDS: 2048 x 16B chunks ----
    {
        unsigned short* Bnc = const_cast<unsigned short*>(Bt);
        for (int c = threadIdx.x; c < 2048; c += 128) {
            const int row = c >> 5;
            const int co  = (c & 31) * 8;
            unsigned short* g = Bnc + (size_t)(n0 + row) * D + co;
            unsigned short* l = &ldsB[row * BROW + co];
#if HAVE_ASYNC_LDS
            __builtin_amdgcn_global_load_async_to_lds_b128(
                (__attribute__((address_space(1))) v4i_async*)g,
                (__attribute__((address_space(3))) v4i_async*)l, 0, 0);
#else
            *(uint4*)l = *(const uint4*)g;
#endif
        }
#if HAVE_ASYNC_LDS
#if __has_builtin(__builtin_amdgcn_s_wait_asynccnt)
        __builtin_amdgcn_s_wait_asynccnt(0);
#else
        asm volatile("s_wait_asynccnt 0x0" ::: "memory");
#endif
#endif
        __syncthreads();
    }

    // ---- per-wave tiles (clamped so all waves reach the barrier; stores guarded) ----
    const int t0r = blockIdx.x * 8 + wave * 2;
    const int t1r = t0r + 1;
    const int t0  = (t0r < ntiles) ? t0r : (ntiles - 1);
    const int t1  = (t1r < ntiles) ? t1r : (ntiles - 1);
    const int m0  = t0 << 4;
    const int m1  = t1 << 4;

    const int ncol = lane & 15;
    const int kbA  = (lane < 16) ? 0 : 16;      // A row per lane, contiguous K halves
    const int kbB  = (lane < 16) ? 0 : 16;      // B column per lane, contiguous K halves

    const unsigned short* arow0 = A + (size_t)(m0 + ncol) * D;
    const unsigned short* arow1 = A + (size_t)(m1 + ncol) * D;
    const unsigned short* lrow[4];
#pragma unroll
    for (int j = 0; j < 4; ++j)
        lrow[j] = &ldsB[(16 * j + ncol) * BROW + kbB];

    const v8f vzero = {0.f, 0.f, 0.f, 0.f, 0.f, 0.f, 0.f, 0.f};
    v8f acc0[4] = {vzero, vzero, vzero, vzero};
    v8f acc1[4] = {vzero, vzero, vzero, vzero};

#pragma unroll
    for (int k0 = 0; k0 < D; k0 += 32) {
        Frag32B fa0, fa1;
        fa0.u[0] = *(const uint4*)(arow0 + k0 + kbA);
        fa0.u[1] = *(const uint4*)(arow0 + k0 + 8 + kbA);
        fa1.u[0] = *(const uint4*)(arow1 + k0 + kbA);
        fa1.u[1] = *(const uint4*)(arow1 + k0 + 8 + kbA);
#pragma unroll
        for (int j = 0; j < 4; ++j) {
            Frag32B fb;
            fb.u[0] = *(const uint4*)(lrow[j] + k0);       // ds_load_b128
            fb.u[1] = *(const uint4*)(lrow[j] + k0 + 8);
            acc0[j] = __builtin_amdgcn_wmma_f32_16x16x32_bf16(
                false, fa0.v, false, fb.v, (short)0, acc0[j], false, false);
            acc1[j] = __builtin_amdgcn_wmma_f32_16x16x32_bf16(
                false, fa1.v, false, fb.v, (short)0, acc1[j], false, false);
        }
    }

    const int rowoff = (lane < 16) ? 0 : 8;
#pragma unroll
    for (int j = 0; j < 4; ++j) {
        const int col = n0 + 16 * j + ncol;
        const float bb = bias ? bias[col] : 0.0f;
        if (t0r < ntiles) {
            float* cp = C + (size_t)(m0 + rowoff) * D + col;
#pragma unroll
            for (int r = 0; r < 8; ++r) cp[(size_t)r * D] = acc0[j][r] + bb;
        }
        if (t1r < ntiles) {
            float* cp = C + (size_t)(m1 + rowoff) * D + col;
#pragma unroll
            for (int r = 0; r < 8; ++r) cp[(size_t)r * D] = acc1[j][r] + bb;
        }
    }
}

// ---------- edge kernels (wave32 per edge) ----------
__global__ __launch_bounds__(256)
void edge_logits_k(const float* __restrict__ q, const float* __restrict__ k,
                   const int* __restrict__ ei, int E, float* __restrict__ logits)
{
    int w = (int)((blockIdx.x * (unsigned)blockDim.x + threadIdx.x) >> 5);
    int lane = threadIdx.x & 31;
    if (w >= E) return;
    int s = ei[w], dv = ei[E + w];
    const float4* qd = (const float4*)(q + (size_t)dv * D);
    const float4* ks = (const float4*)(k + (size_t)s * D);
    float4 a = qd[lane],      b = ks[lane];
    float acc = a.x * b.x + a.y * b.y + a.z * b.z + a.w * b.w;
    a = qd[lane + 32];        b = ks[lane + 32];
    acc += a.x * b.x + a.y * b.y + a.z * b.z + a.w * b.w;
#pragma unroll
    for (int off = 16; off; off >>= 1) acc += __shfl_xor(acc, off, 32);
    if (lane == 0) logits[w] = acc * 0.0625f;   // 1/sqrt(256)
}

__global__ void edge_max_k(const float* __restrict__ logits, const int* __restrict__ ei,
                           int E, unsigned* __restrict__ mkey)
{
    int e = blockIdx.x * blockDim.x + threadIdx.x;
    if (e < E) atomicMax(&mkey[ei[E + e]], fkey(logits[e]));
}

__global__ void edge_expsum_k(float* __restrict__ logits, const int* __restrict__ ei, int E,
                              const unsigned* __restrict__ mkey, float* __restrict__ ssum,
                              float* __restrict__ deg)
{
    int e = blockIdx.x * blockDim.x + threadIdx.x;
    if (e >= E) return;
    int dv = ei[E + e];
    float ev = __expf(logits[e] - unkey(mkey[dv]));
    logits[e] = ev;
    atomicAdd(&ssum[dv], ev);
    atomicAdd(&deg[dv], 1.0f);
}

__global__ __launch_bounds__(256)
void edge_scatter_v_k(const float* __restrict__ ev, const float* __restrict__ ssum,
                      const float* __restrict__ v, const int* __restrict__ ei, int E,
                      float* __restrict__ agg)
{
    int w = (int)((blockIdx.x * (unsigned)blockDim.x + threadIdx.x) >> 5);
    int lane = threadIdx.x & 31;
    if (w >= E) return;
    int s = ei[w], dv = ei[E + w];
    float wt = ev[w] / ssum[dv];
    const float4* vp = (const float4*)(v + (size_t)s * D);
    float* ar = agg + (size_t)dv * D;
    float4 a = vp[lane];
    atomicAdd(ar + 4 * lane + 0, wt * a.x); atomicAdd(ar + 4 * lane + 1, wt * a.y);
    atomicAdd(ar + 4 * lane + 2, wt * a.z); atomicAdd(ar + 4 * lane + 3, wt * a.w);
    float4 b = vp[lane + 32];
    int o = 4 * (lane + 32);
    atomicAdd(ar + o + 0, wt * b.x); atomicAdd(ar + o + 1, wt * b.y);
    atomicAdd(ar + o + 2, wt * b.z); atomicAdd(ar + o + 3, wt * b.w);
}

__global__ __launch_bounds__(256)
void edge_scatter_h_k(const float* __restrict__ h, const int* __restrict__ ei, int E,
                      float* __restrict__ snb)
{
    int w = (int)((blockIdx.x * (unsigned)blockDim.x + threadIdx.x) >> 5);
    int lane = threadIdx.x & 31;
    if (w >= E) return;
    int s = ei[w], dv = ei[E + w];
    const float4* hp = (const float4*)(h + (size_t)s * D);
    float* ar = snb + (size_t)dv * D;
    float4 a = hp[lane];
    atomicAdd(ar + 4 * lane + 0, a.x); atomicAdd(ar + 4 * lane + 1, a.y);
    atomicAdd(ar + 4 * lane + 2, a.z); atomicAdd(ar + 4 * lane + 3, a.w);
    float4 b = hp[lane + 32];
    int o = 4 * (lane + 32);
    atomicAdd(ar + o + 0, b.x); atomicAdd(ar + o + 1, b.y);
    atomicAdd(ar + o + 2, b.z); atomicAdd(ar + o + 3, b.w);
}

// ---------- node epilogues ----------
__global__ void node_h0_k(const float* __restrict__ agg, const float* __restrict__ skip,
                          float* __restrict__ h, unsigned short* __restrict__ hb, int n)
{
    int i = blockIdx.x * blockDim.x + threadIdx.x;
    if (i >= n) return;
    float v = agg[i] + skip[i];
    v = fmaxf(v, 0.0f);
    h[i] = v;
    hb[i] = f2bf(v);
}

__global__ void node_mean_cast_k(const float* __restrict__ snb, const float* __restrict__ deg,
                                 unsigned short* __restrict__ ab, int n)
{
    int i = blockIdx.x * blockDim.x + threadIdx.x;
    if (i >= n) return;
    float dg = fmaxf(deg[i >> 8], 1.0f);
    ab[i] = f2bf(snb[i] / dg);
}

__global__ void node_sage_ep_k(const float* __restrict__ z1, const float* __restrict__ z2,
                               const float* __restrict__ hprev,
                               const float* __restrict__ gamma, const float* __restrict__ beta,
                               const float* __restrict__ alpha,
                               float* __restrict__ hout, unsigned short* __restrict__ hb, int n)
{
    int i = blockIdx.x * blockDim.x + threadIdx.x;
    if (i >= n) return;
    int d = i & (D - 1);
    float al = 1.0f / (1.0f + __expf(-alpha[0]));
    float z = (z1[i] + z2[i]) * BN_SCALE * gamma[d] + beta[d];
    float v = fmaxf(al * z + (1.0f - al) * hprev[i], 0.0f);
    hout[i] = v;
    if (hb) hb[i] = f2bf(v);
}

// ---------- host launcher ----------
extern "C" void kernel_launch(void* const* d_in, const int* in_sizes, int n_in,
                              void* d_out, int out_size, void* d_ws, size_t ws_size,
                              hipStream_t stream)
{
    (void)n_in; (void)out_size; (void)ws_size;
    const float* x     = (const float*)d_in[0];
    const int*   ei    = (const int*)d_in[1];
    const float* Wq    = (const float*)d_in[2];
    const float* bq    = (const float*)d_in[3];
    const float* Wk    = (const float*)d_in[4];
    const float* bk    = (const float*)d_in[5];
    const float* Wv    = (const float*)d_in[6];
    const float* bv    = (const float*)d_in[7];
    const float* Wsk   = (const float*)d_in[8];
    const float* bs    = (const float*)d_in[9];
    const float* Wl    = (const float*)d_in[10];
    const float* bl    = (const float*)d_in[11];
    const float* Wr    = (const float*)d_in[12];
    const float* gamma = (const float*)d_in[13];
    const float* beta  = (const float*)d_in[14];
    const float* alpha = (const float*)d_in[15];

    const int N  = in_sizes[0] / D;          // 10000
    const int E  = in_sizes[1] / 2;          // 320000
    const int ND = N * D;
    const int NT = N / 16;                   // 625 row tiles

    // --- carve workspace ---
    char* ws = (char*)d_ws;
    size_t off = 0;
    auto carve = [&](size_t bytes) -> void* {
        void* p = ws + off;
        off = (off + bytes + 255) & ~(size_t)255;
        return p;
    };
    float* bQ  = (float*)carve((size_t)ND * 4);   // q  -> agg -> z1
    float* bK  = (float*)carve((size_t)ND * 4);   // k  -> hA
    float* bV  = (float*)carve((size_t)ND * 4);   // v  -> z2
    float* bS  = (float*)carve((size_t)ND * 4);   // skip -> sum_nb
    float* bH2 = (float*)carve((size_t)ND * 4);   // hB
    unsigned short* xb = (unsigned short*)carve((size_t)ND * 2);
    unsigned short* ab = (unsigned short*)carve((size_t)ND * 2);
    unsigned short* hb = (unsigned short*)carve((size_t)ND * 2);
    unsigned short* WT[10];
    for (int i = 0; i < 10; ++i) WT[i] = (unsigned short*)carve((size_t)D * D * 2);
    float*    ev   = (float*)carve((size_t)E * 4);
    float*    ssum = (float*)carve((size_t)N * 4);
    float*    deg  = (float*)carve((size_t)N * 4);
    unsigned* mkey = (unsigned*)carve((size_t)N * 4);

    const dim3 blk256(256);
    const dim3 gND((ND + 255) / 256);
    const dim3 gN((N + 255) / 256);
    const dim3 gE((E + 255) / 256);
    const dim3 gEw((E + 7) / 8);                 // wave-per-edge, 8 waves/block
    const dim3 gGemm((NT + 7) / 8, D / 64);      // 4 waves/block, 2 tiles/wave
    const dim3 blk128(128);

    // 1) pack operands: x -> bf16, weights -> transposed bf16
    cast_bf16_k<<<gND, blk256, 0, stream>>>(x, xb, ND);
    transpose_cast_k<<<dim3(D), dim3(D), 0, stream>>>(Wq,  WT[0]);
    transpose_cast_k<<<dim3(D), dim3(D), 0, stream>>>(Wk,  WT[1]);
    transpose_cast_k<<<dim3(D), dim3(D), 0, stream>>>(Wv,  WT[2]);
    transpose_cast_k<<<dim3(D), dim3(D), 0, stream>>>(Wsk, WT[3]);
    for (int i = 0; i < 3; ++i) {
        transpose_cast_k<<<dim3(D), dim3(D), 0, stream>>>(Wl + (size_t)i * D * D, WT[4 + i]);
        transpose_cast_k<<<dim3(D), dim3(D), 0, stream>>>(Wr + (size_t)i * D * D, WT[7 + i]);
    }

    // 2) q/k/v/skip projections (WMMA bf16)
    gemm_bf16_wmma_k<<<gGemm, blk128, 0, stream>>>(xb, WT[0], bq, bQ, NT);
    gemm_bf16_wmma_k<<<gGemm, blk128, 0, stream>>>(xb, WT[1], bk, bK, NT);
    gemm_bf16_wmma_k<<<gGemm, blk128, 0, stream>>>(xb, WT[2], bv, bV, NT);
    gemm_bf16_wmma_k<<<gGemm, blk128, 0, stream>>>(xb, WT[3], bs, bS, NT);

    // 3) attention: logits, segment max/softmax, weighted scatter
    init_nodes_k<<<gN, blk256, 0, stream>>>(mkey, ssum, deg, N);
    edge_logits_k<<<gEw, blk256, 0, stream>>>(bQ, bK, ei, E, ev);
    edge_max_k<<<gE, blk256, 0, stream>>>(ev, ei, E, mkey);
    edge_expsum_k<<<gE, blk256, 0, stream>>>(ev, ei, E, mkey, ssum, deg);
    zero_f32_k<<<gND, blk256, 0, stream>>>(bQ, ND);                       // bQ := agg
    edge_scatter_v_k<<<gEw, blk256, 0, stream>>>(ev, ssum, bV, ei, E, bQ);
    node_h0_k<<<gND, blk256, 0, stream>>>(bQ, bS, bK, hb, ND);            // h0 in bK

    // 4) SAGE layers
    float* hcur = bK;
    for (int i = 0; i < 3; ++i) {
        float* hnext = (i == 0) ? bH2 : (i == 1) ? bK : (float*)d_out;
        zero_f32_k<<<gND, blk256, 0, stream>>>(bS, ND);                   // bS := sum_nb
        edge_scatter_h_k<<<gEw, blk256, 0, stream>>>(hcur, ei, E, bS);
        node_mean_cast_k<<<gND, blk256, 0, stream>>>(bS, deg, ab, ND);
        gemm_bf16_wmma_k<<<gGemm, blk128, 0, stream>>>(ab, WT[4 + i], bl + (size_t)i * D, bQ, NT); // z1
        gemm_bf16_wmma_k<<<gGemm, blk128, 0, stream>>>(hb, WT[7 + i], (const float*)nullptr, bV, NT); // z2
        node_sage_ep_k<<<gND, blk256, 0, stream>>>(bQ, bV, hcur,
                                                   gamma + (size_t)i * D, beta + (size_t)i * D,
                                                   alpha, hnext,
                                                   (i < 2) ? hb : (unsigned short*)nullptr, ND);
        hcur = hnext;
    }
}
